// Gemma2AttentionSdpa_73143293051227
// MI455X (gfx1250) — compile-verified, matrix operand order checked by reading, not run on hardware
//
#include <hip/hip_runtime.h>

// ---------------- problem constants ----------------
#define BATCH   2
#define SEQ     2048
#define HIDDEN  3584
#define NH      16
#define NKV     8
#define HD      256
#define SW      1024
#define CAPF    50.0f
#define SCALEF  0.0625f          // 256^-0.5
#define NTOK    (BATCH*SEQ)      // 4096
#define NQDIM   (NH*HD)          // 4096
#define NKVDIM  (NKV*HD)         // 2048

typedef __attribute__((ext_vector_type(16))) __bf16 bf16x16;
typedef __attribute__((ext_vector_type(8)))  float  f32x8;
typedef unsigned short u16;

// ---------------- small helpers ----------------
__device__ __forceinline__ u16 f2bf(float x) {
    unsigned u = __float_as_uint(x);
    unsigned r = (u + 0x7FFFu + ((u >> 16) & 1u)) >> 16;
    return (u16)r;
}
__device__ __forceinline__ float bf2f(u16 h) {
    return __uint_as_float(((unsigned)h) << 16);
}

__device__ __forceinline__ f32x8 wmma_bf16(bf16x16 a, bf16x16 b, f32x8 c) {
    return __builtin_amdgcn_wmma_f32_16x16x32_bf16(false, a, false, b,
                                                   (short)0, c, false, false);
}

// CDNA5 async DMA: copy 16 bytes global -> LDS, tracked by ASYNCcnt.
// LDS address operand = byte offset within workgroup LDS (generic addr[31:0]).
__device__ __forceinline__ unsigned lds_addr(const void* p) {
    return (unsigned)(size_t)p;
}
__device__ __forceinline__ void async_ld16(const u16* g, unsigned ldsoff) {
    asm volatile("global_load_async_to_lds_b128 %0, %1, off"
                 :: "v"(ldsoff), "v"(g) : "memory");
}
__device__ __forceinline__ void wait_async0() {
    asm volatile("s_wait_asynccnt 0x0" ::: "memory");
}

// Load a 16x32 bf16 A/B fragment from row-major global memory, ld in elements.
//   lanes 0-15 : row = lane,    K = {0..7, 16..23}
//   lanes16-31 : row = lane-16, K = {8..15, 24..31}
__device__ __forceinline__ bf16x16 load_frag(const u16* __restrict__ p, int ld, int lane) {
    int l16 = lane & 15, hi = lane >> 4;
    const u16* q = p + (size_t)l16 * ld + hi * 8;
    union { bf16x16 f; uint4 u[2]; } r;
    r.u[0] = *(const uint4*)(q);
    r.u[1] = *(const uint4*)(q + 16);
    return r.f;
}

// ---------------- conversion kernels ----------------
__global__ void convert_bf16_kernel(const float* __restrict__ in,
                                    u16* __restrict__ out, size_t n) {
    size_t i = (size_t)blockIdx.x * blockDim.x + threadIdx.x;
    if (i < n) out[i] = f2bf(in[i]);
}

// in: K x N row-major f32 ; out: N x K row-major bf16 (i.e. transposed)
__global__ void transpose_bf16_kernel(const float* __restrict__ in,
                                      u16* __restrict__ out, int K, int N) {
    size_t i = (size_t)blockIdx.x * blockDim.x + threadIdx.x;
    size_t total = (size_t)K * N;
    if (i >= total) return;
    int k = (int)(i % K);
    int n = (int)(i / K);
    out[(size_t)n * K + k] = f2bf(in[(size_t)k * N + n]);
}

// ---------------- bf16 WMMA GEMM with async-LDS double buffering ----------
//  C(MxN) = A(MxK) * B(KxN);  A row-major, BT = B^T row-major ([N][K]).
//  Block: 256 threads / 8 waves in 4(M) x 2(N); block tile 128x128, k-step 32.
#define BM 128
#define BN 128
#define BK 32
#define LROW 40   // padded LDS row in ushorts (80B, multiple of 16B)

__global__ __launch_bounds__(256)
void gemm_bf16_kernel(const u16* __restrict__ A, const u16* __restrict__ BT,
                      float* __restrict__ Cf, u16* __restrict__ Cb,
                      int M, int N, int K) {
    __shared__ __align__(16) u16 Abuf[2][BM * LROW];
    __shared__ __align__(16) u16 Bbuf[2][BN * LROW];

    int tid   = threadIdx.x;
    int lane  = tid & 31;
    int wave  = tid >> 5;
    int waveM = wave & 3;          // 0..3  -> 32-row slice
    int waveN = wave >> 2;         // 0..1  -> 64-col slice
    int tileM = blockIdx.y * BM;
    int tileN = blockIdx.x * BN;
    int l16 = lane & 15, hi = lane >> 4;

    // issue async copies for one 128x32 A tile + 128x32 B tile (4 x 16B / thread)
    auto stage = [&](int buf, int kk) {
#pragma unroll
        for (int r = 0; r < 2; ++r) {
            int idx = tid + r * 256;       // 0..511
            int row = idx >> 2;            // 0..127
            int ch  = idx & 3;             // 16B chunk within 64B row
            async_ld16(A  + (size_t)(tileM + row) * K + kk + ch * 8,
                       lds_addr(&Abuf[buf][row * LROW + ch * 8]));
            async_ld16(BT + (size_t)(tileN + row) * K + kk + ch * 8,
                       lds_addr(&Bbuf[buf][row * LROW + ch * 8]));
        }
    };
    // read a 16x32 fragment from a staged LDS tile (rows padded to LROW)
    auto ldsfrag = [&](const u16* base, int row0) -> bf16x16 {
        const u16* q = base + (row0 + l16) * LROW + hi * 8;
        union { bf16x16 f; uint4 u[2]; } r;
        r.u[0] = *(const uint4*)(q);
        r.u[1] = *(const uint4*)(q + 16);
        return r.f;
    };

    f32x8 acc[2][4];
#pragma unroll
    for (int i = 0; i < 2; ++i)
#pragma unroll
        for (int j = 0; j < 4; ++j) acc[i][j] = (f32x8){0,0,0,0,0,0,0,0};

    int nst = K / BK;
    stage(0, 0);
    for (int s = 0; s < nst; ++s) {
        wait_async0();        // this wave's stage-s tile is in LDS
        __syncthreads();      // everyone's tile in LDS; stage s-1 reads done
        if (s + 1 < nst) stage((s + 1) & 1, (s + 1) * BK);   // overlap DMA
        const u16* Ab = &Abuf[s & 1][0];
        const u16* Bb = &Bbuf[s & 1][0];
        bf16x16 a0 = ldsfrag(Ab, waveM * 32);
        bf16x16 a1 = ldsfrag(Ab, waveM * 32 + 16);
#pragma unroll
        for (int j = 0; j < 4; ++j) {
            bf16x16 b = ldsfrag(Bb, waveN * 64 + 16 * j);
            acc[0][j] = wmma_bf16(a0, b, acc[0][j]);
            acc[1][j] = wmma_bf16(a1, b, acc[1][j]);
        }
    }

    int rbase = ((lane >> 4) << 3);
    int c0    = (lane & 15);
#pragma unroll
    for (int i = 0; i < 2; ++i) {
#pragma unroll
        for (int j = 0; j < 4; ++j) {
            int rowg = tileM + waveM * 32 + i * 16 + rbase;
            int colg = tileN + waveN * 64 + 16 * j + c0;
#pragma unroll
            for (int r = 0; r < 8; ++r) {
                size_t idx = (size_t)(rowg + r) * N + colg;
                if (Cf) Cf[idx] = acc[i][j][r];
                else    Cb[idx] = f2bf(acc[i][j][r]);
            }
        }
    }
}

// ---------------- RoPE (interleaved pairs) + head transpose ----------------
// in : [B, L, nheads, HD] bf16   out : [B, nheads, L, HD] bf16
__global__ void rope_kernel(const u16* __restrict__ in, u16* __restrict__ out,
                            int nheads) {
    size_t idx = (size_t)blockIdx.x * blockDim.x + threadIdx.x;
    size_t total = (size_t)BATCH * SEQ * nheads * (HD / 2);
    if (idx >= total) return;
    int d2 = (int)(idx & (HD / 2 - 1));
    size_t rest = idx >> 7;                 // HD/2 == 128
    int h = (int)(rest % nheads); rest /= nheads;
    int l = (int)(rest % SEQ);
    int b = (int)(rest / SEQ);

    float invf = __powf(10000.0f, -(float)(2 * d2) / (float)HD);
    float ang  = (float)l * invf;
    float c = __cosf(ang), s = __sinf(ang);

    size_t src = (((size_t)b * SEQ + l) * nheads + h) * HD + 2 * d2;
    float x0 = bf2f(in[src]), x1 = bf2f(in[src + 1]);
    size_t dst = (((size_t)b * nheads + h) * SEQ + l) * HD + 2 * d2;
    out[dst]     = f2bf(x0 * c - x1 * s);
    out[dst + 1] = f2bf(x1 * c + x0 * s);
}

// in : [B, L, NKV, HD] bf16   out : [B, NKV, HD, L] bf16
__global__ void vtrans_kernel(const u16* __restrict__ in, u16* __restrict__ out) {
    size_t idx = (size_t)blockIdx.x * blockDim.x + threadIdx.x;
    size_t total = (size_t)BATCH * NKV * HD * SEQ;
    if (idx >= total) return;
    int l = (int)(idx % SEQ); size_t rest = idx / SEQ;
    int d = (int)(rest % HD); rest /= HD;
    int h = (int)(rest % NKV);
    int b = (int)(rest / NKV);
    out[idx] = in[(((size_t)b * SEQ + l) * NKV + h) * HD + d];
}

// ---------------- sliding-window flash attention ---------------------------
// Block: 128 threads / 4 waves; wave w owns q rows [qbase+16w, +16).
// K/V for each 32-key step are async-staged into shared LDS (double buffered)
// and reused by all 4 waves.
// Q : [B,NH,L,HD]  K : [B,NKV,L,HD]  Vt : [B,NKV,HD,L]  O : [B,L,NH*HD] bf16
#define KROW 264   // 256 + 8 pad ushorts (528B, multiple of 16B)
#define VROW 40    // 32 + 8 pad ushorts  (80B, multiple of 16B)

__global__ __launch_bounds__(128)
void attn_kernel(const u16* __restrict__ Q, const u16* __restrict__ Kc,
                 const u16* __restrict__ Vt, u16* __restrict__ O) {
    __shared__ __align__(16) u16 Kbuf[2][32 * KROW];   // [key][d]
    __shared__ __align__(16) u16 Vbuf[2][HD * VROW];   // [d][key]
    __shared__ __align__(16) u16 plds[4][16 * 32];     // per-wave P transpose

    int tid  = threadIdx.x;
    int lane = tid & 31;
    int wave = tid >> 5;
    int l16 = lane & 15, hi = lane >> 4;
    int qbase = blockIdx.x * 64;
    int h = blockIdx.y, b = blockIdx.z;
    int q0 = qbase + wave * 16;
    int kvh = h / (NH / NKV);

    const u16* qbaseP = Q  + (((size_t)b * NH  + h)   * SEQ + q0) * HD;
    const u16* kbase  = Kc + (((size_t)b * NKV + kvh) * SEQ) * HD;
    const u16* vbase  = Vt + (((size_t)b * NKV + kvh) * HD) * (size_t)SEQ;

    // stage one 32-key K tile (32x256) + V tile (256x32) : 16 x 16B / thread
    auto stageKV = [&](int buf, int jb) {
#pragma unroll
        for (int r = 0; r < 8; ++r) {
            int idx = tid + r * 128;            // 0..1023
            int krow = idx >> 5, kch = idx & 31;
            async_ld16(kbase + (size_t)(jb + krow) * HD + kch * 8,
                       lds_addr(&Kbuf[buf][krow * KROW + kch * 8]));
            int vd = idx >> 2, vch = idx & 3;
            async_ld16(vbase + (size_t)vd * SEQ + jb + vch * 8,
                       lds_addr(&Vbuf[buf][vd * VROW + vch * 8]));
        }
    };
    auto ldsfrag = [&](const u16* base, int row0, int rowstride, int k0) -> bf16x16 {
        const u16* q = base + (row0 + l16) * rowstride + k0 + hi * 8;
        union { bf16x16 f; uint4 u[2]; } r;
        r.u[0] = *(const uint4*)(q);
        r.u[1] = *(const uint4*)(q + 16);
        return r.f;
    };

    bf16x16 qf[8];
#pragma unroll
    for (int d8 = 0; d8 < 8; ++d8) qf[d8] = load_frag(qbaseP + d8 * 32, HD, lane);

    f32x8 o[16];
#pragma unroll
    for (int n = 0; n < 16; ++n) o[n] = (f32x8){0,0,0,0,0,0,0,0};
    float mrow[8], lrow[8];
#pragma unroll
    for (int e = 0; e < 8; ++e) { mrow[e] = -1e30f; lrow[e] = 0.0f; }

    int jlo = qbase - (SW - 1); if (jlo < 0) jlo = 0; jlo &= ~31;
    int jhi = qbase + 63;                 // inclusive upper key bound for block

    stageKV(0, jlo);
    int sidx = 0;
    for (int jb = jlo; jb <= jhi; jb += 32, ++sidx) {
        wait_async0();                    // this wave's tile copies done
        __syncthreads();                  // all waves' tiles in LDS
        if (jb + 32 <= jhi) stageKV((sidx + 1) & 1, jb + 32);   // overlap DMA
        const u16* Kb = &Kbuf[sidx & 1][0];
        const u16* Vb = &Vbuf[sidx & 1][0];

        f32x8 s[2];
        s[0] = (f32x8){0,0,0,0,0,0,0,0};
        s[1] = (f32x8){0,0,0,0,0,0,0,0};
#pragma unroll
        for (int d8 = 0; d8 < 8; ++d8) {
            s[0] = wmma_bf16(qf[d8], ldsfrag(Kb, 0,  KROW, d8 * 32), s[0]);
            s[1] = wmma_bf16(qf[d8], ldsfrag(Kb, 16, KROW, d8 * 32), s[1]);
        }

        // scale + softcap + band mask ; track per-row max
        float rm[8];
#pragma unroll
        for (int e = 0; e < 8; ++e) rm[e] = -1e30f;
#pragma unroll
        for (int t = 0; t < 2; ++t) {
            int j = jb + 16 * t + l16;
#pragma unroll
            for (int e = 0; e < 8; ++e) {
                int i = q0 + e + hi * 8;
                float v = s[t][e] * SCALEF;
                float z = v * (1.0f / CAPF);
                z = fminf(fmaxf(z, -15.0f), 15.0f);
                float e2 = __expf(2.0f * z);
                v = CAPF * ((e2 - 1.0f) / (e2 + 1.0f));
                bool ok = (j <= i) && (i - j <= SW - 1);
                v = ok ? v : -1e30f;
                s[t][e] = v;
                rm[e] = fmaxf(rm[e], v);
            }
        }
#pragma unroll
        for (int e = 0; e < 8; ++e) {
            float r = rm[e];
            for (int off = 1; off < 16; off <<= 1)
                r = fmaxf(r, __shfl_xor(r, off, 32));
            rm[e] = r;
        }
        float alpha[8];
#pragma unroll
        for (int e = 0; e < 8; ++e) {
            float mn = fmaxf(mrow[e], rm[e]);
            alpha[e] = (mn <= -1e29f) ? 1.0f : __expf(mrow[e] - mn);
            mrow[e] = mn;
        }
        float ls[8];
#pragma unroll
        for (int e = 0; e < 8; ++e) ls[e] = 0.0f;
#pragma unroll
        for (int t = 0; t < 2; ++t)
#pragma unroll
            for (int e = 0; e < 8; ++e) {
                float p = (s[t][e] <= -1e29f) ? 0.0f : __expf(s[t][e] - mrow[e]);
                s[t][e] = p;
                ls[e] += p;
            }
#pragma unroll
        for (int e = 0; e < 8; ++e) {
            float r = ls[e];
            for (int off = 1; off < 16; off <<= 1)
                r += __shfl_xor(r, off, 32);
            lrow[e] = lrow[e] * alpha[e] + r;
        }
#pragma unroll
        for (int n = 0; n < 16; ++n)
#pragma unroll
            for (int e = 0; e < 8; ++e) o[n][e] *= alpha[e];

        // C-layout -> A-fragment layout via per-wave LDS scratch
        // (DS ops from one wave are in-order; no block barrier needed)
        u16* pw = &plds[wave][0];
#pragma unroll
        for (int t = 0; t < 2; ++t)
#pragma unroll
            for (int e = 0; e < 8; ++e)
                pw[(e + hi * 8) * 32 + 16 * t + l16] = f2bf(s[t][e]);
        union { bf16x16 f; uint4 u[2]; } pf;
        const u16* pp = pw + l16 * 32 + hi * 8;
        pf.u[0] = *(const uint4*)pp;
        pf.u[1] = *(const uint4*)(pp + 16);

        // O += P * V
#pragma unroll
        for (int n = 0; n < 16; ++n)
            o[n] = wmma_bf16(pf.f, ldsfrag(Vb, 16 * n, VROW, 0), o[n]);
    }

    float inv[8];
#pragma unroll
    for (int e = 0; e < 8; ++e) inv[e] = (lrow[e] > 0.0f) ? 1.0f / lrow[e] : 0.0f;
#pragma unroll
    for (int n = 0; n < 16; ++n)
#pragma unroll
        for (int e = 0; e < 8; ++e) {
            int i = q0 + e + hi * 8;
            int col = h * HD + 16 * n + l16;
            O[((size_t)b * SEQ + i) * NQDIM + col] = f2bf(o[n][e] * inv[e]);
        }
}

// ---------------- host orchestration ----------------
extern "C" void kernel_launch(void* const* d_in, const int* in_sizes, int n_in,
                              void* d_out, int out_size, void* d_ws, size_t ws_size,
                              hipStream_t stream) {
    (void)in_sizes; (void)n_in; (void)out_size; (void)ws_size;
    const float* x  = (const float*)d_in[0];
    const float* wq = (const float*)d_in[1];
    const float* wk = (const float*)d_in[2];
    const float* wv = (const float*)d_in[3];
    const float* wo = (const float*)d_in[4];
    /* d_in[5] = causal bool mask, applied analytically */
    float* out = (float*)d_out;

    char* ws = (char*)d_ws;
    size_t off = 0;
    auto alloc = [&](size_t bytes) -> void* {
        void* p = ws + off;
        off += (bytes + 255) & ~(size_t)255;
        return p;
    };
    u16* xb   = (u16*)alloc((size_t)NTOK  * HIDDEN * 2);
    u16* wqT  = (u16*)alloc((size_t)NQDIM * HIDDEN * 2);
    u16* wkT  = (u16*)alloc((size_t)NKVDIM* HIDDEN * 2);
    u16* wvT  = (u16*)alloc((size_t)NKVDIM* HIDDEN * 2);
    u16* woT  = (u16*)alloc((size_t)HIDDEN* NQDIM  * 2);
    u16* qtmp = (u16*)alloc((size_t)NTOK  * NQDIM  * 2);   // also reused as ao
    u16* ktmp = (u16*)alloc((size_t)NTOK  * NKVDIM * 2);
    u16* vtmp = (u16*)alloc((size_t)NTOK  * NKVDIM * 2);
    u16* qr   = (u16*)alloc((size_t)NTOK  * NQDIM  * 2);
    u16* kr   = (u16*)alloc((size_t)NTOK  * NKVDIM * 2);
    u16* vt   = (u16*)alloc((size_t)NTOK  * NKVDIM * 2);
    u16* ao   = qtmp;  // safe reuse: rope consumes qtmp before attn writes ao

    // 1. convert / transpose inputs to bf16
    {
        size_t n = (size_t)NTOK * HIDDEN;
        convert_bf16_kernel<<<(unsigned)((n + 255) / 256), 256, 0, stream>>>(x, xb, n);
        size_t nq = (size_t)HIDDEN * NQDIM;
        transpose_bf16_kernel<<<(unsigned)((nq + 255) / 256), 256, 0, stream>>>(wq, wqT, HIDDEN, NQDIM);
        size_t nk = (size_t)HIDDEN * NKVDIM;
        transpose_bf16_kernel<<<(unsigned)((nk + 255) / 256), 256, 0, stream>>>(wk, wkT, HIDDEN, NKVDIM);
        transpose_bf16_kernel<<<(unsigned)((nk + 255) / 256), 256, 0, stream>>>(wv, wvT, HIDDEN, NKVDIM);
        size_t no = (size_t)NQDIM * HIDDEN;
        transpose_bf16_kernel<<<(unsigned)((no + 255) / 256), 256, 0, stream>>>(wo, woT, NQDIM, HIDDEN);
    }

    // 2. QKV projections (async-LDS double-buffered WMMA GEMMs, bf16 out)
    {
        dim3 gq(NQDIM / BN, NTOK / BM);
        gemm_bf16_kernel<<<gq, 256, 0, stream>>>(xb, wqT, nullptr, qtmp, NTOK, NQDIM, HIDDEN);
        dim3 gk(NKVDIM / BN, NTOK / BM);
        gemm_bf16_kernel<<<gk, 256, 0, stream>>>(xb, wkT, nullptr, ktmp, NTOK, NKVDIM, HIDDEN);
        gemm_bf16_kernel<<<gk, 256, 0, stream>>>(xb, wvT, nullptr, vtmp, NTOK, NKVDIM, HIDDEN);
    }

    // 3. RoPE + layout transforms
    {
        size_t npq = (size_t)BATCH * SEQ * NH * (HD / 2);
        rope_kernel<<<(unsigned)((npq + 255) / 256), 256, 0, stream>>>(qtmp, qr, NH);
        size_t npk = (size_t)BATCH * SEQ * NKV * (HD / 2);
        rope_kernel<<<(unsigned)((npk + 255) / 256), 256, 0, stream>>>(ktmp, kr, NKV);
        size_t nv = (size_t)BATCH * NKV * HD * SEQ;
        vtrans_kernel<<<(unsigned)((nv + 255) / 256), 256, 0, stream>>>(vtmp, vt);
    }

    // 4. sliding-window flash attention (shared async-staged K/V)
    {
        dim3 ga(SEQ / 64, NH, BATCH);
        attn_kernel<<<ga, 128, 0, stream>>>(qr, kr, vt, ao);
    }

    // 5. output projection -> f32 d_out
    {
        dim3 go(HIDDEN / BN, NTOK / BM);
        gemm_bf16_kernel<<<go, 256, 0, stream>>>(ao, woT, out, nullptr, NTOK, HIDDEN, NQDIM);
    }
}
